// DeepSwarmLDA_26542897889836
// MI455X (gfx1250) — matrix operand order repaired and checked by reading, not compiled, not run on hardware
//
#include <hip/hip_runtime.h>
#include <hip/hip_bf16.h>
#include <math.h>

typedef __bf16 bf16_t;
typedef __attribute__((ext_vector_type(16))) __bf16 v16bf;
typedef __attribute__((ext_vector_type(8)))  __bf16 v8bf;
typedef __attribute__((ext_vector_type(8)))  float  v8f;

#define DIM_B  1024
#define DIM_NG 2000
#define DIM_L  1000
#define DIM_G  50
#define DIM_C  10
#define KP     1024   // L padded to multiple of 32 for WMMA K
#define N0     100
#define NP     112    // N0 padded to multiple of 16

__device__ __forceinline__ float gelu_exact(float x) {
    return 0.5f * x * (1.0f + erff(x * 0.70710678118654752f));
}

// Stage 1: gather + per-l (Bx50)x(50x10) mini-GEMM -> bf16 A matrix (B*C, KP)
// One block = one l, 256 rows of B. idx/weights/bias staged in LDS.
__global__ void k_gather_lda(const float* __restrict__ X,
                             const int*   __restrict__ gidx,
                             const float* __restrict__ ldaW,
                             const float* __restrict__ ldab,
                             bf16_t*      __restrict__ A)
{
    __shared__ int   s_idx[DIM_G];
    __shared__ float s_w[DIM_C * DIM_G];
    __shared__ float s_b[DIM_C];
    const int l = blockIdx.y;
    const int t = threadIdx.x;
    if (t < DIM_G) s_idx[t] = gidx[l * DIM_G + t];
    for (int i = t; i < DIM_C * DIM_G; i += 256) s_w[i] = ldaW[(size_t)l * DIM_C * DIM_G + i];
    if (t < DIM_C) s_b[t] = ldab[l * DIM_C + t];
    __syncthreads();

    const int b = blockIdx.x * 256 + t;
    float acc[DIM_C];
#pragma unroll
    for (int c = 0; c < DIM_C; ++c) acc[c] = s_b[c];

    const float* xrow = X + (size_t)b * DIM_NG;
    for (int g = 0; g < DIM_G; ++g) {
        float xv = xrow[s_idx[g]];
#pragma unroll
        for (int c = 0; c < DIM_C; ++c) acc[c] += xv * s_w[c * DIM_G + g];
    }
#pragma unroll
    for (int c = 0; c < DIM_C; ++c)
        A[(size_t)(b * DIM_C + c) * KP + l] = (bf16_t)acc[c];
}

// Zero the K-pad columns (l = 1000..1023) of A.
__global__ void k_padA(bf16_t* __restrict__ A)
{
    int i = blockIdx.x * 256 + threadIdx.x;        // over 10240 * 24
    int row = i / (KP - DIM_L);
    int col = DIM_L + i % (KP - DIM_L);
    if (row < DIM_B * DIM_C) A[(size_t)row * KP + col] = (bf16_t)0.0f;
}

// Stage 2: (10240 x 1024) x (1024 x 112) bf16 WMMA GEMM, f32 accum,
// fused +b0 and exact GELU. Block = 8 waves = 128 M-rows x 16 N-cols.
__global__ void k_gemm_wmma(const bf16_t* __restrict__ A,
                            const float*  __restrict__ W0,
                            const float*  __restrict__ b0,
                            float*        __restrict__ H)
{
    // B fragments pre-swizzled: [chunk(32)][lane(32)][e(16)] bf16 = 32 KB
    __shared__ __align__(32) bf16_t sB[32 * 32 * 16];

    const int t     = threadIdx.x;
    const int lane  = t & 31;
    const int w     = t >> 5;          // wave 0..7
    const int tileN = blockIdx.y;      // 0..6
    const int blkM  = blockIdx.x;      // 0..79

    // Cooperative fill: lane `ln` of chunk `ch` wants, for element e,
    // B[k = ch*32 + 16*(e>=8) + 8*(ln>=16) + (e&7)][n = ln&15] = W0^T, zero-padded.
    for (int i = 0; i < 64; ++i) {
        int idx = t + i * 256;
        int e   = idx & 15;
        int ln  = (idx >> 4) & 31;
        int ch  = idx >> 9;
        int k   = ch * 32 + ((e >> 3) << 4) + ((ln >> 4) << 3) + (e & 7);
        int n   = tileN * 16 + (ln & 15);
        float v = (k < DIM_L && n < N0) ? W0[(size_t)n * DIM_L + k] : 0.0f;
        sB[idx] = (bf16_t)v;
    }
    __syncthreads();

    const int half    = lane >> 4;     // K-half select for A fragment
    const int mrow    = lane & 15;
    const int rowBase = blkM * 128 + w * 16;
    const bf16_t* arow = A + (size_t)(rowBase + mrow) * KP;

    v8f acc = {};
    for (int ch = 0; ch < 32; ++ch) {
        const int k0 = ch * 32;
        // A fragment: two contiguous 16-byte chunks per lane (ISA 16-bit A layout)
        v8bf alo = *(const v8bf*)(arow + k0 + 8 * half);
        v8bf ahi = *(const v8bf*)(arow + k0 + 16 + 8 * half);
        v16bf afrag;
#pragma unroll
        for (int i = 0; i < 8; ++i) { afrag[i] = alo[i]; afrag[i + 8] = ahi[i]; }
        // B fragment: 16 contiguous bf16 per lane from the swizzled LDS stage
        v16bf bfrag = *(const v16bf*)(&sB[(ch * 32 + lane) * 16]);

        acc = __builtin_amdgcn_wmma_f32_16x16x32_bf16(
            /*neg_a=*/false, afrag, /*neg_b=*/false, bfrag,
            /*c_mod=*/(short)0, acc, /*reuse_a=*/false, /*reuse_b=*/false);
    }

    // D layout: element r -> M = r + 8*half, N = lane&15
    const int n = tileN * 16 + (lane & 15);
    if (n < N0) {
        const float bias = b0[n];
#pragma unroll
        for (int r = 0; r < 8; ++r) {
            int m   = rowBase + r + 8 * half;
            float x = acc[r] + bias;
            H[(size_t)m * N0 + n] = gelu_exact(x);
        }
    }
}

// Stage 3: per-row head: gelu(h@W1^T+b1) @ W2^T + b2 -> out (10240 floats)
__global__ void k_head(const float* __restrict__ H,
                       const float* __restrict__ W1, const float* __restrict__ b1,
                       const float* __restrict__ W2, const float* __restrict__ b2,
                       float* __restrict__ out)
{
    __shared__ float sW1[10 * 100];
    __shared__ float sb1[10];
    __shared__ float sW2[10];
    const int t = threadIdx.x;
    for (int i = t; i < 1000; i += 256) sW1[i] = W1[i];
    if (t < 10) { sb1[t] = b1[t]; sW2[t] = W2[t]; }
    __syncthreads();

    const int r = blockIdx.x * 256 + t;
    const float* h = H + (size_t)r * N0;
    float s[10];
#pragma unroll
    for (int j = 0; j < 10; ++j) s[j] = sb1[j];
    for (int k = 0; k < N0; ++k) {
        float hv = h[k];
#pragma unroll
        for (int j = 0; j < 10; ++j) s[j] += hv * sW1[j * N0 + k];
    }
    float o = b2[0];
#pragma unroll
    for (int j = 0; j < 10; ++j) o += gelu_exact(s[j]) * sW2[j];
    out[r] = o;
}

extern "C" void kernel_launch(void* const* d_in, const int* in_sizes, int n_in,
                              void* d_out, int out_size, void* d_ws, size_t ws_size,
                              hipStream_t stream)
{
    const float* X    = (const float*)d_in[0];
    const int*   gidx = (const int*)  d_in[1];
    const float* ldaW = (const float*)d_in[2];
    const float* ldab = (const float*)d_in[3];
    const float* W0   = (const float*)d_in[4];
    const float* b0   = (const float*)d_in[5];
    const float* W1   = (const float*)d_in[6];
    const float* b1   = (const float*)d_in[7];
    const float* W2   = (const float*)d_in[8];
    const float* b2   = (const float*)d_in[9];
    float* out = (float*)d_out;

    char* ws = (char*)d_ws;
    bf16_t* A = (bf16_t*)ws;                                   // 10240*1024*2 = 20 MB
    float*  H = (float*)(ws + (size_t)DIM_B * DIM_C * KP * 2); // 10240*100*4  =  4 MB

    k_gather_lda<<<dim3(DIM_B / 256, DIM_L), 256, 0, stream>>>(X, gidx, ldaW, ldab, A);
    k_padA<<<(DIM_B * DIM_C * (KP - DIM_L) + 255) / 256, 256, 0, stream>>>(A);
    k_gemm_wmma<<<dim3(DIM_B * DIM_C / 128, NP / 16), 256, 0, stream>>>(A, W0, b0, H);
    k_head<<<DIM_B * DIM_C / 256, 256, 0, stream>>>(H, W1, b1, W2, b2, out);
}
